// LinearAttention_37366215475355
// MI455X (gfx1250) — compile-verified, matrix operand order checked by reading, not run on hardware
//
#include <hip/hip_runtime.h>
#include <hip/hip_bf16.h>
#include <stdint.h>

// ---------------------------------------------------------------------------
// LinearAttention forward for MI455X (gfx1250), wave32 + WMMA f16->f32.
// b=16, c=256, n=4096, heads=4, d=32, hidden=128.
// ---------------------------------------------------------------------------

typedef __attribute__((ext_vector_type(16))) _Float16 v16h;
typedef __attribute__((ext_vector_type(8)))  _Float16 v8h;
typedef __attribute__((ext_vector_type(8)))  float    v8f;

#define BATCH 16
#define CH    256
#define NTOK  4096
#define HID   128
#define QKVR  384
#define NHEAD 4
#define DHEAD 32
#define NMEM  4

__device__ __forceinline__ v8f wmma_f16(v16h a, v16h b, v8f c) {
  // v_wmma_f32_16x16x32_f16 : D = A(16x32) * B(32x16) + C(16x16 f32)
  return __builtin_amdgcn_wmma_f32_16x16x32_f16(
      /*neg_a=*/false, a, /*neg_b=*/false, b,
      /*c_mod=*/(short)0, c, /*reuse_a=*/false, /*reuse_b=*/false);
}

// Load a 16x32 (MxK) f16 fragment from an LDS tile stored row-major with
// `stride` halves per row.  Per ISA 7.12.2: lane L<16 holds row M=L with
// K = {0..7, 16..23}; lane L>=16 holds row M=L-16 with K = {8..15, 24..31}.
// Each half-run is 8 contiguous halves -> one b128 LDS load.
__device__ __forceinline__ v16h load_frag(const _Float16* base, int lane, int stride) {
  const int r  = lane & 15;
  const int k0 = (lane >> 4) << 3;       // 0 or 8
  const _Float16* p = base + r * stride;
  v8h lo = *(const v8h*)(p + k0);
  v8h hi = *(const v8h*)(p + 16 + k0);
  v16h f;
#pragma unroll
  for (int i = 0; i < 8; ++i) { f[i] = lo[i]; f[i + 8] = hi[i]; }
  return f;
}

// ---------------------------------------------------------------------------
// K0: fold norm_g into qkv weight, convert to f16.
// ---------------------------------------------------------------------------
__global__ void prep_kernel(const float* __restrict__ w_qkv,
                            const float* __restrict__ norm_g,
                            _Float16* __restrict__ wqkv_h) {
  int i = blockIdx.x * 256 + threadIdx.x;
  if (i < QKVR * CH) wqkv_h[i] = (_Float16)(w_qkv[i] * norm_g[i & (CH - 1)]);
}

// ---------------------------------------------------------------------------
// K1: per-(b,n) channel inverse norm, with sqrt(c)=16 folded in.
// ---------------------------------------------------------------------------
__global__ void invnorm_kernel(const float* __restrict__ x,
                               float* __restrict__ inv_norm) {
  int pos = blockIdx.x * 256 + threadIdx.x;     // 0 .. 65535
  int b = pos >> 12, n = pos & (NTOK - 1);
  const float* col = x + (size_t)b * CH * NTOK + n;
  float s = 0.f;
  for (int c = 0; c < CH; ++c) { float v = col[(size_t)c * NTOK]; s += v * v; }
  inv_norm[pos] = 16.0f / fmaxf(sqrtf(s), 1e-12f);
}

// ---------------------------------------------------------------------------
// K2: QKV GEMM, WMMA.  Per workgroup: 384x32 output tile, K=256.
// Fused: q-softmax over d (per head, per column) * SCALE -> qs_t (b,n,128) f16
//        k raw f32 (b,128,n);  v f16 (b,128,n).
// ---------------------------------------------------------------------------
__global__ __launch_bounds__(256) void qkv_kernel(
    const float* __restrict__ x, const _Float16* __restrict__ wqkv_h,
    const float* __restrict__ inv_norm,
    _Float16* __restrict__ qs_t, float* __restrict__ kraw,
    _Float16* __restrict__ vh) {
  // LDS: stage phase  : A 384x32 f16 (24576 B) + B 32x32 f16 (2048 B)
  //      epilog phase : out 384x33 f32 (50688 B)   (reuses same memory)
  __shared__ __align__(16) char smem[QKVR * 33 * 4];
  _Float16* lds_a = (_Float16*)smem;
  _Float16* lds_b = (_Float16*)(smem + QKVR * 32 * 2);
  float*    lds_o = (float*)smem;

  const int tid  = threadIdx.x;
  const int wave = tid >> 5, lane = tid & 31;
  const int n0   = blockIdx.x * 32;
  const int b    = blockIdx.y;

  v8f acc[3][2];
#pragma unroll
  for (int i = 0; i < 3; ++i)
#pragma unroll
    for (int j = 0; j < 2; ++j)
#pragma unroll
      for (int r = 0; r < 8; ++r) acc[i][j][r] = 0.f;

  const uint32_t* wsrc = (const uint32_t*)wqkv_h;
  uint32_t*       adst = (uint32_t*)lds_a;

  for (int ks = 0; ks < CH / 32; ++ks) {
    const int K0 = ks * 32;
    __syncthreads();
    // stage A: 384x32 halves == 384x16 dwords
    for (int idx = tid; idx < QKVR * 16; idx += 256) {
      int r = idx >> 4, cp = idx & 15;
      adst[idx] = wsrc[r * (CH / 2) + (K0 >> 1) + cp];
    }
    // stage B (transposed [n][c]): x * inv_norm -> f16
#pragma unroll
    for (int i = 0; i < 4; ++i) {
      int idx = tid + i * 256;
      int nl = idx & 31, cl = idx >> 5;
      float xv = x[((size_t)b * CH + (K0 + cl)) * NTOK + n0 + nl];
      lds_b[nl * 32 + cl] = (_Float16)(xv * inv_norm[b * NTOK + n0 + nl]);
    }
    __syncthreads();
    v16h bf0 = load_frag(lds_b, lane, 32);
    v16h bf1 = load_frag(lds_b + 16 * 32, lane, 32);
#pragma unroll
    for (int i = 0; i < 3; ++i) {
      int R = (wave * 3 + i) * 16;
      v16h af = load_frag(lds_a + R * 32, lane, 32);
      acc[i][0] = wmma_f16(af, bf0, acc[i][0]);
      acc[i][1] = wmma_f16(af, bf1, acc[i][1]);
    }
  }
  __syncthreads();
  // dump accumulators: C layout -> row = R + (lane>=16?8:0) + r, col = lane&15
#pragma unroll
  for (int i = 0; i < 3; ++i) {
    int Rb = (wave * 3 + i) * 16 + ((lane >> 4) << 3);
#pragma unroll
    for (int j = 0; j < 2; ++j) {
      int cc = j * 16 + (lane & 15);
#pragma unroll
      for (int r = 0; r < 8; ++r) lds_o[(Rb + r) * 33 + cc] = acc[i][j][r];
    }
  }
  __syncthreads();
  // q rows 0..127: softmax over d per (head, column), * 32^-0.5
  if (tid < 128) {
    const float SCALE = 0.17677669529663687f;
    int h = tid >> 5, col = tid & 31;
    float m = -1e30f;
#pragma unroll
    for (int d = 0; d < DHEAD; ++d) m = fmaxf(m, lds_o[(h * 32 + d) * 33 + col]);
    float e[DHEAD], s = 0.f;
#pragma unroll
    for (int d = 0; d < DHEAD; ++d) {
      e[d] = __expf(lds_o[(h * 32 + d) * 33 + col] - m);
      s += e[d];
    }
    float inv = SCALE / s;
    _Float16* dst = qs_t + ((size_t)b * NTOK + n0 + col) * HID + h * 32;
#pragma unroll
    for (int d = 0; d < DHEAD; ++d) dst[d] = (_Float16)(e[d] * inv);
  }
  // k rows 128..255 raw f32; v rows 256..383 f16
  for (int idx = tid; idx < HID * 32; idx += 256) {
    int r = idx >> 5, col = idx & 31;
    kraw[((size_t)b * HID + r) * NTOK + n0 + col] = lds_o[(HID + r) * 33 + col];
  }
  for (int idx = tid; idx < HID * 32; idx += 256) {
    int r = idx >> 5, col = idx & 31;
    vh[((size_t)b * HID + r) * NTOK + n0 + col] =
        (_Float16)lds_o[(2 * HID + r) * 33 + col];
  }
}

// ---------------------------------------------------------------------------
// K3: per-row (b,h,d) max and sum(exp) over n (incl. 4 memory tokens).
// ---------------------------------------------------------------------------
__global__ __launch_bounds__(256) void kstats_kernel(
    const float* __restrict__ kraw, const float* __restrict__ mem_kv,
    float* __restrict__ rowmax, float* __restrict__ rowsum) {
  __shared__ float red[256];
  const int rg = blockIdx.x;          // 0..2047  == b*128 + h*32 + d
  const int tid = threadIdx.x;
  const int hd = rg & (HID - 1);
  const float* row = kraw + (size_t)rg * NTOK;
  float m = -1e30f;
  for (int n = tid; n < NTOK; n += 256) m = fmaxf(m, row[n]);
  if (tid < NMEM) m = fmaxf(m, mem_kv[hd * NMEM + tid]);   // mem_kv[0]
  red[tid] = m; __syncthreads();
  for (int s = 128; s > 0; s >>= 1) {
    if (tid < s) red[tid] = fmaxf(red[tid], red[tid + s]);
    __syncthreads();
  }
  m = red[0]; __syncthreads();
  float s = 0.f;
  for (int n = tid; n < NTOK; n += 256) s += __expf(row[n] - m);
  if (tid < NMEM) s += __expf(mem_kv[hd * NMEM + tid] - m);
  red[tid] = s; __syncthreads();
  for (int st = 128; st > 0; st >>= 1) {
    if (tid < st) red[tid] += red[tid + st];
    __syncthreads();
  }
  if (tid == 0) { rowmax[rg] = m; rowsum[rg] = red[0]; }
}

// ---------------------------------------------------------------------------
// K4: context[b,h,d,e] = sum_n softmax_k[d,n] * v[e,n]  (incl. mem tokens).
// One workgroup per (b,h); LDS-tiled over n.
// ---------------------------------------------------------------------------
__global__ __launch_bounds__(256) void context_kernel(
    const float* __restrict__ kraw, const _Float16* __restrict__ vh,
    const float* __restrict__ mem_kv, const float* __restrict__ rowmax,
    const float* __restrict__ rowsum, float* __restrict__ context) {
  __shared__ float kch[32][65];
  __shared__ float vch[32][65];
  __shared__ float rmax[32];
  const int bh = blockIdx.x, b = bh >> 2, h = bh & 3;
  const int tid = threadIdx.x;
  const int d = tid >> 3, e0 = (tid & 7) * 4;
  const size_t rowbase = ((size_t)b * HID + h * 32) * NTOK;
  const int statbase = b * HID + h * 32;
  if (tid < 32) rmax[tid] = rowmax[statbase + tid];
  float acc[4] = {0.f, 0.f, 0.f, 0.f};
  for (int chunk = 0; chunk < NTOK; chunk += 64) {
    __syncthreads();
#pragma unroll
    for (int i = 0; i < 8; ++i) {
      int idx = tid + i * 256;
      int r = idx >> 6, c = idx & 63;
      float kv = kraw[rowbase + (size_t)r * NTOK + chunk + c];
      kch[r][c] = __expf(kv - rmax[r]);
      vch[r][c] = (float)vh[rowbase + (size_t)r * NTOK + chunk + c];
    }
    __syncthreads();
#pragma unroll 8
    for (int j = 0; j < 64; ++j) {
      float kx = kch[d][j];
      acc[0] += kx * vch[e0 + 0][j];
      acc[1] += kx * vch[e0 + 1][j];
      acc[2] += kx * vch[e0 + 2][j];
      acc[3] += kx * vch[e0 + 3][j];
    }
  }
  // memory tokens: mem_kv[0]=keys at offset 0, mem_kv[1]=values at offset 512
#pragma unroll
  for (int j = 0; j < NMEM; ++j) {
    float kx = __expf(mem_kv[(h * 32 + d) * NMEM + j] - rmax[d]);
#pragma unroll
    for (int i = 0; i < 4; ++i)
      acc[i] += kx * mem_kv[NHEAD * DHEAD * NMEM + (h * 32 + e0 + i) * NMEM + j];
  }
  float invs = 1.0f / rowsum[statbase + d];
#pragma unroll
  for (int i = 0; i < 4; ++i)
    context[(((size_t)b * NHEAD + h) * DHEAD + d) * DHEAD + e0 + i] = acc[i] * invs;
}

// ---------------------------------------------------------------------------
// K5: W2[b,o,h*32+d] = sum_e w_out[o,h*32+e] * context[b,h,d,e]  -> f16
// (collapses "out = context^T @ q" into the final projection GEMM)
// ---------------------------------------------------------------------------
__global__ __launch_bounds__(256) void w2_kernel(
    const float* __restrict__ w_out, const float* __restrict__ context,
    _Float16* __restrict__ W2h) {
  __shared__ float ctxT[NHEAD][DHEAD][33];   // [h][e][d]
  const int b = blockIdx.x, tid = threadIdx.x;
  for (int idx = tid; idx < NHEAD * DHEAD * DHEAD; idx += 256) {
    int h = idx >> 10, rem = idx & 1023, d = rem >> 5, e = rem & 31;
    ctxT[h][e][d] = context[(size_t)b * 4096 + idx];
  }
  __syncthreads();
  const int o = tid;
  _Float16* dst = W2h + ((size_t)b * CH + o) * HID;
  float acc[DHEAD];
#pragma unroll 1
  for (int h = 0; h < NHEAD; ++h) {
#pragma unroll
    for (int d = 0; d < DHEAD; ++d) acc[d] = 0.f;
    for (int e = 0; e < DHEAD; ++e) {
      float w = w_out[o * HID + h * 32 + e];
#pragma unroll
      for (int d = 0; d < DHEAD; ++d) acc[d] += w * ctxT[h][e][d];
    }
#pragma unroll
    for (int d = 0; d < DHEAD; ++d) dst[h * 32 + d] = (_Float16)acc[d];
  }
}

// ---------------------------------------------------------------------------
// K6: Y = W2[b] @ qs  + b_out, then RMSNorm over the 256 output channels.
// WMMA; one workgroup owns a full 256-row x 32-col tile so the channel
// norm is done entirely in LDS.
// ---------------------------------------------------------------------------
__global__ __launch_bounds__(256) void out_kernel(
    const _Float16* __restrict__ W2h, const _Float16* __restrict__ qs_t,
    const float* __restrict__ b_out, const float* __restrict__ out_g,
    float* __restrict__ out) {
  __shared__ __align__(16) char smem[CH * 33 * 4];
  _Float16* lds_a = (_Float16*)smem;
  _Float16* lds_b = (_Float16*)(smem + CH * 32 * 2);
  float*    lds_y = (float*)smem;
  __shared__ float colinv[32];

  const int tid = threadIdx.x;
  const int wave = tid >> 5, lane = tid & 31;
  const int n0 = blockIdx.x * 32;
  const int b  = blockIdx.y;

  v8f acc[2][2];
#pragma unroll
  for (int i = 0; i < 2; ++i)
#pragma unroll
    for (int j = 0; j < 2; ++j)
#pragma unroll
      for (int r = 0; r < 8; ++r) acc[i][j][r] = 0.f;

  const uint32_t* asrc = (const uint32_t*)W2h;
  const uint32_t* bsrc = (const uint32_t*)qs_t;
  uint32_t* adst = (uint32_t*)lds_a;
  uint32_t* bdst = (uint32_t*)lds_b;

  for (int ks = 0; ks < HID / 32; ++ks) {
    const int K0 = ks * 32;
    __syncthreads();
    for (int idx = tid; idx < CH * 16; idx += 256) {      // A: 256x32 halves
      int r = idx >> 4, cp = idx & 15;
      adst[idx] = asrc[((size_t)b * CH + r) * (HID / 2) + (K0 >> 1) + cp];
    }
#pragma unroll
    for (int i = 0; i < 2; ++i) {                          // B: 32x32 halves
      int idx = tid + i * 256;
      int n = idx >> 4, cp = idx & 15;
      bdst[idx] = bsrc[((size_t)b * NTOK + n0 + n) * (HID / 2) + (K0 >> 1) + cp];
    }
    __syncthreads();
    v16h bf0 = load_frag(lds_b, lane, 32);
    v16h bf1 = load_frag(lds_b + 16 * 32, lane, 32);
#pragma unroll
    for (int i = 0; i < 2; ++i) {
      int R = (wave * 2 + i) * 16;
      v16h af = load_frag(lds_a + R * 32, lane, 32);
      acc[i][0] = wmma_f16(af, bf0, acc[i][0]);
      acc[i][1] = wmma_f16(af, bf1, acc[i][1]);
    }
  }
  __syncthreads();
#pragma unroll
  for (int i = 0; i < 2; ++i) {
    int Rb = (wave * 2 + i) * 16 + ((lane >> 4) << 3);
#pragma unroll
    for (int j = 0; j < 2; ++j) {
      int cc = j * 16 + (lane & 15);
#pragma unroll
      for (int r = 0; r < 8; ++r) lds_y[(Rb + r) * 33 + cc] = acc[i][j][r];
    }
  }
  __syncthreads();
  if (tid < 32) {
    float s = 0.f;
    for (int o = 0; o < CH; ++o) {
      float v = lds_y[o * 33 + tid] + b_out[o];
      s += v * v;
    }
    colinv[tid] = 16.0f / fmaxf(sqrtf(s), 1e-12f);
  }
  __syncthreads();
  for (int idx = tid; idx < CH * 32; idx += 256) {
    int o = idx >> 5, col = idx & 31;
    float v = (lds_y[o * 33 + col] + b_out[o]) * colinv[col] * out_g[o];
    out[((size_t)b * CH + o) * NTOK + n0 + col] = v;
  }
}

// ---------------------------------------------------------------------------
extern "C" void kernel_launch(void* const* d_in, const int* in_sizes, int n_in,
                              void* d_out, int out_size, void* d_ws, size_t ws_size,
                              hipStream_t stream) {
  (void)in_sizes; (void)n_in; (void)out_size; (void)ws_size;
  const float* x        = (const float*)d_in[0];   // (16,256,64,64)
  const float* norm_g   = (const float*)d_in[1];   // (256)
  const float* w_qkv    = (const float*)d_in[2];   // (384,256)
  const float* mem_kv   = (const float*)d_in[3];   // (2,4,32,4)
  const float* w_out    = (const float*)d_in[4];   // (256,128)
  const float* b_out    = (const float*)d_in[5];   // (256)
  const float* out_g    = (const float*)d_in[6];   // (256)
  float* out = (float*)d_out;                      // (16,256,64,64)

  char* ws = (char*)d_ws;
  size_t off = 0;
  auto take = [&](size_t bytes) -> char* {
    char* p = ws + off;
    off = (off + bytes + 255) & ~(size_t)255;
    return p;
  };
  _Float16* wqkv_h   = (_Float16*)take((size_t)QKVR * CH * 2);           // 192 KB
  float*    inv_norm = (float*)   take((size_t)BATCH * NTOK * 4);        // 256 KB
  _Float16* qs_t     = (_Float16*)take((size_t)BATCH * NTOK * HID * 2);  // 16 MB
  float*    kraw     = (float*)   take((size_t)BATCH * HID * NTOK * 4);  // 32 MB
  _Float16* vh       = (_Float16*)take((size_t)BATCH * HID * NTOK * 2);  // 16 MB
  float*    rowmax   = (float*)   take((size_t)BATCH * HID * 4);
  float*    rowsum   = (float*)   take((size_t)BATCH * HID * 4);
  float*    context  = (float*)   take((size_t)BATCH * NHEAD * DHEAD * DHEAD * 4);
  _Float16* W2h      = (_Float16*)take((size_t)BATCH * CH * HID * 2);    // 1 MB

  prep_kernel<<<QKVR, 256, 0, stream>>>(w_qkv, norm_g, wqkv_h);
  invnorm_kernel<<<BATCH * NTOK / 256, 256, 0, stream>>>(x, inv_norm);
  qkv_kernel<<<dim3(NTOK / 32, BATCH), 256, 0, stream>>>(
      x, wqkv_h, inv_norm, qs_t, kraw, vh);
  kstats_kernel<<<BATCH * HID, 256, 0, stream>>>(kraw, mem_kv, rowmax, rowsum);
  context_kernel<<<BATCH * NHEAD, 256, 0, stream>>>(
      kraw, vh, mem_kv, rowmax, rowsum, context);
  w2_kernel<<<BATCH, 256, 0, stream>>>(w_out, context, W2h);
  out_kernel<<<dim3(NTOK / 32, BATCH), 256, 0, stream>>>(
      W2h, qs_t, b_out, out_g, out);
}
